// MIC_RWKV_4217657884748
// MI455X (gfx1250) — matrix operand order repaired and checked by reading, not compiled, not run on hardware
//
#include <hip/hip_runtime.h>
#include <cstdint>
#include <cstddef>

// ---------------------------------------------------------------------------
// RWKV block for MI455X (gfx1250): bf16 WMMA GEMMs + fused epilogues,
// chunk-parallel WKV scan, LDS-reduction LayerNorm, token-shift mix.
// ---------------------------------------------------------------------------

typedef __attribute__((ext_vector_type(16))) __bf16 bf16x16;
typedef __attribute__((ext_vector_type(8)))  __bf16 bf16x8;
typedef __attribute__((ext_vector_type(8)))  float  f32x8;

#define BB 8
#define TT 2048
#define CC 1024
#define HH 4096
#define MM (BB*TT)
#define NCH 16              // WKV scan chunks
#define CHLEN (TT/NCH)      // 128 timesteps per chunk

static __device__ __forceinline__ uint16_t f2bfu(float f) {
  uint32_t u = __builtin_bit_cast(uint32_t, f);
  u += 0x7fffu + ((u >> 16) & 1u);   // round-to-nearest-even
  return (uint16_t)(u >> 16);
}
static __device__ __forceinline__ __bf16 f2bf(float f) {
  uint16_t h = f2bfu(f);
  return __builtin_bit_cast(__bf16, h);
}

// ------------------------- fp32 -> bf16 conversion (x4) --------------------
__global__ __launch_bounds__(256) void cvt_f32_bf16_v4(const float4* __restrict__ s,
                                                       ushort4* __restrict__ d, int n4) {
  int i = blockIdx.x * 256 + threadIdx.x;
  if (i < n4) {
    const float4 f = s[i];
    ushort4 o;
    o.x = f2bfu(f.x); o.y = f2bfu(f.y); o.z = f2bfu(f.z); o.w = f2bfu(f.w);
    d[i] = o;
  }
}

// ------------------------------- LayerNorm ---------------------------------
// one block (256 threads) per row of length C=1024; 4 floats per thread
__global__ __launch_bounds__(256) void ln_rows(const float* __restrict__ x,
                                               const float* __restrict__ g,
                                               const float* __restrict__ b,
                                               float* __restrict__ y) {
  __shared__ float s1[256];
  __shared__ float s2[256];
  const int row = blockIdx.x, tid = threadIdx.x;
  const float4 xv = reinterpret_cast<const float4*>(x + (size_t)row * CC)[tid];
  s1[tid] = xv.x + xv.y + xv.z + xv.w;
  s2[tid] = xv.x * xv.x + xv.y * xv.y + xv.z * xv.z + xv.w * xv.w;
  __syncthreads();
  for (int o = 128; o > 0; o >>= 1) {
    if (tid < o) { s1[tid] += s1[tid + o]; s2[tid] += s2[tid + o]; }
    __syncthreads();
  }
  const float mean = s1[0] * (1.0f / CC);
  const float var  = s2[0] * (1.0f / CC) - mean * mean;
  const float rs   = rsqrtf(var + 1e-5f);
  const float4 gv = reinterpret_cast<const float4*>(g)[tid];
  const float4 bv = reinterpret_cast<const float4*>(b)[tid];
  float4 o4;
  o4.x = (xv.x - mean) * rs * gv.x + bv.x;
  o4.y = (xv.y - mean) * rs * gv.y + bv.y;
  o4.z = (xv.z - mean) * rs * gv.z + bv.z;
  o4.w = (xv.w - mean) * rs * gv.w + bv.w;
  reinterpret_cast<float4*>(y + (size_t)row * CC)[tid] = o4;
}

// ----------------------------- token-shift mix (x4) ------------------------
// xm = x*tm + shift(x,-1)*(1-tm) + (c<C/2 ? shift(x,-1) : shift(x,+1))*cm
__global__ __launch_bounds__(256) void mix_shift(const float* __restrict__ x,
                                                 const float* __restrict__ tm,
                                                 const float* __restrict__ cm,
                                                 __bf16* __restrict__ out) {
  const size_t e = ((size_t)blockIdx.x * 256 + threadIdx.x) * 4;
  const int c = (int)(e & (CC - 1));            // multiple of 4
  const size_t bt = e >> 10;                    // C == 1024
  const int t = (int)(bt & (TT - 1));           // T == 2048
  const float4 zero = {0.f, 0.f, 0.f, 0.f};
  const float4 x0  = *reinterpret_cast<const float4*>(x + e);
  const float4 xm1 = (t > 0)      ? *reinterpret_cast<const float4*>(x + e - CC) : zero;
  const float4 xp1 = (t < TT - 1) ? *reinterpret_cast<const float4*>(x + e + CC) : zero;
  const float4 tm4 = *reinterpret_cast<const float4*>(tm + c);
  const float4 cm4 = *reinterpret_cast<const float4*>(cm + c);
  const bool lo = (c < CC / 2);                 // 512 is 4-aligned: uniform over the 4
  const float4 xc = lo ? xm1 : xp1;
  ushort4 o;
  o.x = f2bfu(x0.x * tm4.x + xm1.x * (1.0f - tm4.x) + xc.x * cm4.x);
  o.y = f2bfu(x0.y * tm4.y + xm1.y * (1.0f - tm4.y) + xc.y * cm4.y);
  o.z = f2bfu(x0.z * tm4.z + xm1.z * (1.0f - tm4.z) + xc.z * cm4.z);
  o.w = f2bfu(x0.w * tm4.w + xm1.w * (1.0f - tm4.w) + xc.w * cm4.w);
  reinterpret_cast<ushort4*>(out)[e >> 2] = o;
}

// --------------------------- WKV chunked scan ------------------------------
// Linear recurrence (a,b) <- (ed*a + k*v, ed*b + k) composed over chunks:
// a_out = ed^L * a_in + A_partial.  Three phases for 16x more parallelism.

// phase 1: per-(batch,channel,chunk) partial sums. gid = [chunk][batch][c]
__global__ __launch_bounds__(256) void wkv_partial(const float* __restrict__ kk,
                                                   const float* __restrict__ vv,
                                                   const float* __restrict__ td,
                                                   float* __restrict__ Pa,
                                                   float* __restrict__ Pb) {
  const int gid = blockIdx.x * 256 + threadIdx.x;   // B*C*NCH
  const int c = gid & (CC - 1);
  const int rest = gid >> 10;
  const int batch = rest & (BB - 1);
  const int chunk = rest >> 3;                      // BB == 8
  const float ed = expf(-expf(td[c]));
  float a = 0.0f, b = 0.0f;
  size_t idx = ((size_t)batch * TT + (size_t)chunk * CHLEN) * CC + c;
  for (int t = 0; t < CHLEN; ++t, idx += CC) {
    const float kt = kk[idx], vt = vv[idx];
    a = ed * a + kt * vt;
    b = ed * b + kt;
  }
  Pa[gid] = a; Pb[gid] = b;
}

// phase 2: tiny 16-step scan over chunk partials -> incoming state per chunk
__global__ __launch_bounds__(256) void wkv_combine(const float* __restrict__ Pa,
                                                   const float* __restrict__ Pb,
                                                   const float* __restrict__ td,
                                                   float* __restrict__ Sa,
                                                   float* __restrict__ Sb) {
  const int bc = blockIdx.x * 256 + threadIdx.x;    // B*C
  const int c = bc & (CC - 1);
  const float edp = expf(-expf(td[c]) * (float)CHLEN);   // ed^CHLEN
  float a = 0.0f, b = 0.0f;
#pragma unroll
  for (int ch = 0; ch < NCH; ++ch) {
    const int p = ch * (BB * CC) + bc;
    Sa[p] = a; Sb[p] = b;
    a = edp * a + Pa[p];
    b = edp * b + Pb[p];
  }
}

// phase 3: replay chunk from incoming state, emit g = sigmoid(r) * y (bf16)
__global__ __launch_bounds__(256) void wkv_emit(const float* __restrict__ kk,
                                                const float* __restrict__ vv,
                                                const float* __restrict__ sr,
                                                const float* __restrict__ td,
                                                const float* __restrict__ tf,
                                                const float* __restrict__ Sa,
                                                const float* __restrict__ Sb,
                                                __bf16* __restrict__ g) {
  const int gid = blockIdx.x * 256 + threadIdx.x;   // B*C*NCH
  const int c = gid & (CC - 1);
  const int rest = gid >> 10;
  const int batch = rest & (BB - 1);
  const int chunk = rest >> 3;
  const float ed = expf(-expf(td[c]));
  const float eu = expf(tf[c]);
  float a = Sa[gid], b = Sb[gid];
  size_t idx = ((size_t)batch * TT + (size_t)chunk * CHLEN) * CC + c;
  for (int t = 0; t < CHLEN; ++t, idx += CC) {
    const float kt = kk[idx], vt = vv[idx];
    const float y = (a + eu * kt * vt) / (b + eu * kt);
    g[idx] = f2bf(sr[idx] * y);
    a = ed * a + kt * vt;
    b = ed * b + kt;
  }
}

// ------------------------------ WMMA GEMM ----------------------------------
// Y[M,N] = X[M,K] @ W[N,K]^T   (bf16 in, f32 accumulate, fused epilogue)
// block: 256 threads = 8 waves; wave tile 32x64 (2 M-tiles x 4 N-tiles);
// block tile 128x128. grid = (N/128, M/128).
enum { EPI_NONE = 0, EPI_EXPMIN = 1, EPI_SIG = 2, EPI_RES = 3,
       EPI_RELU2_BF = 4, EPI_FFNOUT = 5 };

union AFrag { bf16x16 v; bf16x8 h[2]; };

template<int EPI>
__global__ __launch_bounds__(256) void gemm_bf16w(const __bf16* __restrict__ X,
                                                  const __bf16* __restrict__ W,
                                                  int K, int N,
                                                  const float* __restrict__ res,
                                                  const float* __restrict__ scale,
                                                  float* __restrict__ outF,
                                                  __bf16* __restrict__ outH) {
  const int tid   = threadIdx.x;
  const int lidx  = tid & 15;          // lane % 16
  const int lhalf = (tid >> 4) & 1;    // lane half within wave32
  const int wave  = tid >> 5;          // 0..7
  const int mBase = blockIdx.y * 128 + (wave & 3) * 32;
  const int nBase = blockIdx.x * 128 + (wave >> 2) * 64;

  const f32x8 zf = {0.f, 0.f, 0.f, 0.f, 0.f, 0.f, 0.f, 0.f};
  f32x8 acc[2][4];
#pragma unroll
  for (int i = 0; i < 2; ++i)
#pragma unroll
    for (int j = 0; j < 4; ++j) acc[i][j] = zf;

  // A: lane holds row mBase+lidx (+16 for second tile); K chunks matching B
  //    layout: lanes 0-15 -> K {0-15}, lanes 16-31 -> K {16-31}
  const __bf16* xrow0 = X + (size_t)(mBase + lidx) * K;
  const __bf16* xrow1 = X + (size_t)(mBase + 16 + lidx) * K;
  // B: lane holds column nBase+j*16+lidx (= row of W), 16 contiguous K at
  //    offset 16*lhalf -> a single 32-byte load per fragment
  const __bf16* wrow[4];
#pragma unroll
  for (int j = 0; j < 4; ++j)
    wrow[j] = W + (size_t)(nBase + j * 16 + lidx) * K + 16 * lhalf;

  for (int k0 = 0; k0 < K; k0 += 32) {
    // speculative WGP-scope prefetch of the streaming X operand one line
    // ahead (locality 3 -> SCOPE_WGP: pulls into all cache levels on miss)
    __builtin_prefetch(xrow0 + k0 + 128, 0, 3);
    __builtin_prefetch(xrow1 + k0 + 128, 0, 3);
    AFrag a0, a1;
    a0.h[0] = *reinterpret_cast<const bf16x8*>(xrow0 + k0 + 16 * lhalf);
    a0.h[1] = *reinterpret_cast<const bf16x8*>(xrow0 + k0 + 8 + 16 * lhalf);
    a1.h[0] = *reinterpret_cast<const bf16x8*>(xrow1 + k0 + 16 * lhalf);
    a1.h[1] = *reinterpret_cast<const bf16x8*>(xrow1 + k0 + 8 + 16 * lhalf);
    bf16x16 bv[4];
#pragma unroll
    for (int j = 0; j < 4; ++j)
      bv[j] = *reinterpret_cast<const bf16x16*>(wrow[j] + k0);
#pragma unroll
    for (int j = 0; j < 4; ++j) {
      acc[0][j] = __builtin_amdgcn_wmma_f32_16x16x32_bf16(
          false, a0.v, false, bv[j], (short)0, acc[0][j], false, false);
      acc[1][j] = __builtin_amdgcn_wmma_f32_16x16x32_bf16(
          false, a1.v, false, bv[j], (short)0, acc[1][j], false, false);
    }
  }

  // D layout: VGPR j -> row j + 8*lhalf; lane lidx -> column
#pragma unroll
  for (int im = 0; im < 2; ++im) {
#pragma unroll
    for (int in = 0; in < 4; ++in) {
      const int col = nBase + in * 16 + lidx;
#pragma unroll
      for (int j = 0; j < 8; ++j) {
        const int row = mBase + im * 16 + j + 8 * lhalf;
        const size_t idx = (size_t)row * N + col;
        const float v = acc[im][in][j];
        if constexpr (EPI == EPI_NONE) {
          outF[idx] = v;
        } else if constexpr (EPI == EPI_EXPMIN) {
          outF[idx] = expf(fminf(v, 60.0f));
        } else if constexpr (EPI == EPI_SIG) {
          outF[idx] = 1.0f / (1.0f + expf(-v));
        } else if constexpr (EPI == EPI_RES) {
          outF[idx] = res[idx] + v;
        } else if constexpr (EPI == EPI_RELU2_BF) {
          const float t = fmaxf(v, 0.0f);
          outH[idx] = f2bf(t * t);
        } else {  // EPI_FFNOUT: out = x2 + sigmoid(r)*kv
          outF[idx] = res[idx] + scale[idx] * v;
        }
      }
    }
  }
}

// ---------------------------------------------------------------------------
extern "C" void kernel_launch(void* const* d_in, const int* in_sizes, int n_in,
                              void* d_out, int out_size, void* d_ws, size_t ws_size,
                              hipStream_t stream) {
  (void)in_sizes; (void)n_in; (void)out_size; (void)ws_size;
  const float* x       = (const float*)d_in[0];
  const float* ln1_g   = (const float*)d_in[1];
  const float* ln1_b   = (const float*)d_in[2];
  const float* ln2_g   = (const float*)d_in[3];
  const float* ln2_b   = (const float*)d_in[4];
  const float* att_tm  = (const float*)d_in[5];
  const float* att_cm  = (const float*)d_in[6];
  const float* tdec    = (const float*)d_in[7];
  const float* tfirst  = (const float*)d_in[8];
  const float* Wk      = (const float*)d_in[9];
  const float* Wv      = (const float*)d_in[10];
  const float* Wr      = (const float*)d_in[11];
  const float* Wo      = (const float*)d_in[12];
  const float* ffn_tm  = (const float*)d_in[13];
  const float* ffn_cm  = (const float*)d_in[14];
  const float* Fk      = (const float*)d_in[15];
  const float* Fv      = (const float*)d_in[16];
  const float* Fr      = (const float*)d_in[17];

  char* ws = (char*)d_ws;
  size_t off = 0;
  auto take = [&](size_t bytes) -> void* {
    void* p = ws + off;
    off += (bytes + 255) & ~(size_t)255;
    return p;
  };
  // bf16 weights (~26 MB)
  __bf16* Wk16 = (__bf16*)take((size_t)CC * CC * 2);
  __bf16* Wv16 = (__bf16*)take((size_t)CC * CC * 2);
  __bf16* Wr16 = (__bf16*)take((size_t)CC * CC * 2);
  __bf16* Wo16 = (__bf16*)take((size_t)CC * CC * 2);
  __bf16* Fr16 = (__bf16*)take((size_t)CC * CC * 2);
  __bf16* Fk16 = (__bf16*)take((size_t)HH * CC * 2);
  __bf16* Fv16 = (__bf16*)take((size_t)CC * HH * 2);
  // activations
  float*  A    = (float*)take((size_t)MM * CC * 4);   // LN1(x) (residual)
  __bf16* Xm   = (__bf16*)take((size_t)MM * CC * 2);  // mixed input (bf16)
  float*  Kb   = (float*)take((size_t)MM * CC * 4);   // k, later reused as x_att
  float*  Vb   = (float*)take((size_t)MM * CC * 4);   // v, later reused as LN2 out
  float*  Rb   = (float*)take((size_t)MM * CC * 4);   // sigmoid(r)
  __bf16* G16  = (__bf16*)take((size_t)MM * CC * 2);  // sigmoid(r)*wkv (bf16)
  __bf16* H16  = (__bf16*)take((size_t)MM * HH * 2);  // relu(k)^2 hidden (bf16)
  // WKV scan scratch (~2 MB)
  float*  Pa   = (float*)take((size_t)NCH * BB * CC * 4);
  float*  Pb   = (float*)take((size_t)NCH * BB * CC * 4);
  float*  Sa   = (float*)take((size_t)NCH * BB * CC * 4);
  float*  Sb   = (float*)take((size_t)NCH * BB * CC * 4);

  const int nCC = CC * CC, nHC = HH * CC;
  const dim3 blk(256);
  const dim3 gCvt1((nCC / 4 + 255) / 256), gCvt4((nHC / 4 + 255) / 256);
  const dim3 gLN(MM);
  const dim3 gMix(((size_t)MM * CC / 4) / 256);
  const dim3 gGemmC(CC / 128, MM / 128);   // N=1024
  const dim3 gGemmH(HH / 128, MM / 128);   // N=4096
  const dim3 gScan((BB * CC * NCH) / 256); // 512 blocks
  const dim3 gComb((BB * CC) / 256);       // 32 blocks

  // 1) weights -> bf16
  cvt_f32_bf16_v4<<<gCvt1, blk, 0, stream>>>((const float4*)Wk, (ushort4*)Wk16, nCC / 4);
  cvt_f32_bf16_v4<<<gCvt1, blk, 0, stream>>>((const float4*)Wv, (ushort4*)Wv16, nCC / 4);
  cvt_f32_bf16_v4<<<gCvt1, blk, 0, stream>>>((const float4*)Wr, (ushort4*)Wr16, nCC / 4);
  cvt_f32_bf16_v4<<<gCvt1, blk, 0, stream>>>((const float4*)Wo, (ushort4*)Wo16, nCC / 4);
  cvt_f32_bf16_v4<<<gCvt1, blk, 0, stream>>>((const float4*)Fr, (ushort4*)Fr16, nCC / 4);
  cvt_f32_bf16_v4<<<gCvt4, blk, 0, stream>>>((const float4*)Fk, (ushort4*)Fk16, nHC / 4);
  cvt_f32_bf16_v4<<<gCvt4, blk, 0, stream>>>((const float4*)Fv, (ushort4*)Fv16, nHC / 4);

  // 2) x1 = LN1(x); xm = mix(x1)
  ln_rows<<<gLN, blk, 0, stream>>>(x, ln1_g, ln1_b, A);
  mix_shift<<<gMix, blk, 0, stream>>>(A, att_tm, att_cm, Xm);

  // 3) k = exp(min(xm@Wk^T,60)); v = xm@Wv^T; sr = sigmoid(xm@Wr^T)
  gemm_bf16w<EPI_EXPMIN><<<gGemmC, blk, 0, stream>>>(Xm, Wk16, CC, CC, nullptr, nullptr, Kb, nullptr);
  gemm_bf16w<EPI_NONE  ><<<gGemmC, blk, 0, stream>>>(Xm, Wv16, CC, CC, nullptr, nullptr, Vb, nullptr);
  gemm_bf16w<EPI_SIG   ><<<gGemmC, blk, 0, stream>>>(Xm, Wr16, CC, CC, nullptr, nullptr, Rb, nullptr);

  // 4) chunk-parallel WKV scan; g = sigmoid(r) * wkv
  wkv_partial<<<gScan, blk, 0, stream>>>(Kb, Vb, tdec, Pa, Pb);
  wkv_combine<<<gComb, blk, 0, stream>>>(Pa, Pb, tdec, Sa, Sb);
  wkv_emit   <<<gScan, blk, 0, stream>>>(Kb, Vb, Rb, tdec, tfirst, Sa, Sb, G16);

  // 5) x_att = x1 + g @ Wo^T   (write into Kb)
  gemm_bf16w<EPI_RES><<<gGemmC, blk, 0, stream>>>(G16, Wo16, CC, CC, A, nullptr, Kb, nullptr);

  // 6) x2 = LN2(x_att) (into Vb); xm2 = mix(x2)
  ln_rows<<<gLN, blk, 0, stream>>>(Kb, ln2_g, ln2_b, Vb);
  mix_shift<<<gMix, blk, 0, stream>>>(Vb, ffn_tm, ffn_cm, Xm);

  // 7) h = relu(xm2@Fk^T)^2 (bf16); sr2 = sigmoid(xm2@Fr^T)
  gemm_bf16w<EPI_RELU2_BF><<<gGemmH, blk, 0, stream>>>(Xm, Fk16, CC, HH, nullptr, nullptr, nullptr, H16);
  gemm_bf16w<EPI_SIG     ><<<gGemmC, blk, 0, stream>>>(Xm, Fr16, CC, CC, nullptr, nullptr, Rb, nullptr);

  // 8) out = x2 + sr2 * (h @ Fv^T)
  gemm_bf16w<EPI_FFNOUT><<<gGemmC, blk, 0, stream>>>(H16, Fv16, HH, CC, Vb, Rb, (float*)d_out, nullptr);
}
